// GRU4Rec_83562883711202
// MI455X (gfx1250) — compile-verified, matrix operand order checked by reading, not run on hardware
//
#include <hip/hip_runtime.h>
#include <hip/hip_bf16.h>

// ---------------------------------------------------------------------------
// GRU (B=256, T=256, C=1024) on gfx1250.
// Persistent single-kernel recurrence with device-wide software barriers;
// all GEMMs via v_wmma_f32_16x16x32_bf16 with register double-buffered
// fragment loads (overlap L2 latency with WMMA issue inside one wave).
// ---------------------------------------------------------------------------

#define Bdim 256
#define Tdim 256
#define Cdim 1024
#define NBLOCKS 64          // persistent blocks (co-resident; tiny vs MI455X WGP count)
#define NTHREADS 128        // 4 waves of 32

typedef __attribute__((ext_vector_type(8)))  __bf16 v8bf;
typedef __attribute__((ext_vector_type(16))) __bf16 v16bf;
typedef __attribute__((ext_vector_type(8)))  float  v8f;

// --------------------------- fragment loader -------------------------------
// 16-bit A 16x32 layout (ISA 7.12.2): lanes 0-15: row M=lane, K=[0..7]+[16..23];
// lanes 16-31: same rows, K=[8..15]+[24..31].  B (32x16 from row-major W,
// B[k][n]=W[n][k]) mirrors this with N in place of M -> same loader on W rows.
__device__ __forceinline__ v16bf load_frag(const __bf16* __restrict__ row,
                                           int k0, int hi8) {
    v8bf lo = *(const v8bf*)(row + k0 + hi8);
    v8bf hi = *(const v8bf*)(row + k0 + 16 + hi8);
    return __builtin_shufflevector(lo, hi, 0, 1, 2, 3, 4, 5, 6, 7,
                                           8, 9, 10, 11, 12, 13, 14, 15);
}

__device__ __forceinline__ void wmma8(v8f acc[2][4], v16bf a0, v16bf a1,
                                      v16bf b0, v16bf b1, v16bf b2, v16bf b3) {
    acc[0][0] = __builtin_amdgcn_wmma_f32_16x16x32_bf16(false, a0, false, b0, (short)0, acc[0][0], false, false);
    acc[1][0] = __builtin_amdgcn_wmma_f32_16x16x32_bf16(false, a1, false, b0, (short)0, acc[1][0], false, false);
    acc[0][1] = __builtin_amdgcn_wmma_f32_16x16x32_bf16(false, a0, false, b1, (short)0, acc[0][1], false, false);
    acc[1][1] = __builtin_amdgcn_wmma_f32_16x16x32_bf16(false, a1, false, b1, (short)0, acc[1][1], false, false);
    acc[0][2] = __builtin_amdgcn_wmma_f32_16x16x32_bf16(false, a0, false, b2, (short)0, acc[0][2], false, false);
    acc[1][2] = __builtin_amdgcn_wmma_f32_16x16x32_bf16(false, a1, false, b2, (short)0, acc[1][2], false, false);
    acc[0][3] = __builtin_amdgcn_wmma_f32_16x16x32_bf16(false, a0, false, b3, (short)0, acc[0][3], false, false);
    acc[1][3] = __builtin_amdgcn_wmma_f32_16x16x32_bf16(false, a1, false, b3, (short)0, acc[1][3], false, false);
}

// 32x64 strip: acc[mi][ni], rows m0+16*mi.., cols n0+16*ni..  Double-buffered:
// next chunk's 2 A + 4 B fragment loads are issued before this chunk's 8 WMMAs.
__device__ __forceinline__ void gemm_seg32(v8f acc[2][4],
                                           const __bf16* __restrict__ Abase, long lda,
                                           const __bf16* __restrict__ Bbase, long ldb,
                                           int m0, int n0, int K, int lane) {
    const int l16 = lane & 15;
    const int hi8 = (lane >> 4) << 3;
    const __bf16* arow0 = Abase + (long)(m0 + l16) * lda;
    const __bf16* arow1 = arow0 + 16 * lda;
    const __bf16* brow  = Bbase + (long)(n0 + l16) * ldb;

    v16bf a0 = load_frag(arow0, 0, hi8);
    v16bf a1 = load_frag(arow1, 0, hi8);
    v16bf b0 = load_frag(brow,            0, hi8);
    v16bf b1 = load_frag(brow + 16 * ldb, 0, hi8);
    v16bf b2 = load_frag(brow + 32 * ldb, 0, hi8);
    v16bf b3 = load_frag(brow + 48 * ldb, 0, hi8);

#pragma unroll 2
    for (int k0 = 0; k0 < K - 32; k0 += 32) {
        const int kn = k0 + 32;
        __builtin_prefetch(arow0 + k0 + 512, 0, 1);     // global_prefetch_b8
        v16bf na0 = load_frag(arow0, kn, hi8);
        v16bf na1 = load_frag(arow1, kn, hi8);
        v16bf nb0 = load_frag(brow,            kn, hi8);
        v16bf nb1 = load_frag(brow + 16 * ldb, kn, hi8);
        v16bf nb2 = load_frag(brow + 32 * ldb, kn, hi8);
        v16bf nb3 = load_frag(brow + 48 * ldb, kn, hi8);
        wmma8(acc, a0, a1, b0, b1, b2, b3);
        a0 = na0; a1 = na1; b0 = nb0; b1 = nb1; b2 = nb2; b3 = nb3;
    }
    wmma8(acc, a0, a1, b0, b1, b2, b3);                 // peeled last chunk
}

// C/D layout: lane L, VGPR j -> M = j + 8*(L>=16), N = L%16.
#define EPILOGUE_2x4(body)                                                    \
    {                                                                         \
        const int l16 = lane & 15;                                            \
        const int mofs = (lane >> 4) << 3;                                    \
        _Pragma("unroll") for (int mi = 0; mi < 2; ++mi) {                    \
            _Pragma("unroll") for (int ni = 0; ni < 4; ++ni) {                \
                const int col = n0 + 16 * ni + l16;                           \
                _Pragma("unroll") for (int j = 0; j < 8; ++j) {               \
                    const int row = m0 + 16 * mi + mofs + j;                  \
                    const float accv = acc[mi][ni][j];                        \
                    body                                                      \
                }                                                             \
            }                                                                 \
        }                                                                     \
    }

// ------------------------------ parameters ---------------------------------
struct GruParams {
    const __bf16* Xb;                                   // [B,T,C] bf16
    const __bf16 *Wxz, *Whz, *Wxr, *Whr, *Wxh, *Whh, *Whq;
    const float  *bxz, *bhz, *bxr, *bhr, *bxh, *bhh, *bhq;
    float*  Hf;                                         // [B,C] fp32 state
    __bf16* Hb;                                         // [B,C] bf16 state
    float*  Z;                                          // [B,C]
    __bf16* RH;                                         // [B,C]
    float*  Y;                                          // [B,T,C] output
    unsigned* bar_cnt;
    unsigned* bar_gen;
};

// ------------------------ device-wide barrier -------------------------------
__device__ __forceinline__ void gbar(const GruParams& P) {
    __threadfence();                    // release stores to agent scope
    __syncthreads();
    if (threadIdx.x == 0) {
        unsigned g = __hip_atomic_load(P.bar_gen, __ATOMIC_RELAXED,
                                       __HIP_MEMORY_SCOPE_AGENT);
        unsigned prev = __hip_atomic_fetch_add(P.bar_cnt, 1u, __ATOMIC_ACQ_REL,
                                               __HIP_MEMORY_SCOPE_AGENT);
        if (prev == NBLOCKS - 1) {
            __hip_atomic_store(P.bar_cnt, 0u, __ATOMIC_RELAXED,
                               __HIP_MEMORY_SCOPE_AGENT);
            __hip_atomic_fetch_add(P.bar_gen, 1u, __ATOMIC_RELEASE,
                                   __HIP_MEMORY_SCOPE_AGENT);
        } else {
            while (__hip_atomic_load(P.bar_gen, __ATOMIC_ACQUIRE,
                                     __HIP_MEMORY_SCOPE_AGENT) == g)
                __builtin_amdgcn_s_sleep(2);
        }
    }
    __syncthreads();
    __threadfence();                    // acquire
}

// ------------------------------ phase jobs ----------------------------------
// gate job: pre = X_t@Wx^T + H@Wh^T + bx + bh ; z-gate -> Z, r-gate -> bf16(R*H)
__device__ void job_gate(const GruParams& P, int t, int g, int m0, int n0, int lane) {
    const __bf16* Xt = P.Xb + (size_t)t * Cdim;         // row stride T*C
    const __bf16* Wx = g ? P.Wxr : P.Wxz;
    const __bf16* Wh = g ? P.Whr : P.Whz;
    const float*  bx = g ? P.bxr : P.bxz;
    const float*  bh = g ? P.bhr : P.bhz;

    v8f acc[2][4] = {};
    gemm_seg32(acc, Xt,   (long)Tdim * Cdim, Wx, Cdim, m0, n0, Cdim, lane);
    gemm_seg32(acc, P.Hb, Cdim,              Wh, Cdim, m0, n0, Cdim, lane);

    if (g == 0) {
        EPILOGUE_2x4({
            float s = 1.0f / (1.0f + __expf(-(accv + bx[col] + bh[col])));
            P.Z[(long)row * Cdim + col] = s;
        })
    } else {
        EPILOGUE_2x4({
            float s = 1.0f / (1.0f + __expf(-(accv + bx[col] + bh[col])));
            P.RH[(long)row * Cdim + col] =
                (__bf16)(s * P.Hf[(long)row * Cdim + col]);
        })
    }
}

// update job: H~ = tanh(X_t@Wxh^T + RH@Whh^T + b); H = Z*H + (1-Z)*H~
__device__ void job_update(const GruParams& P, int t, int m0, int n0, int lane) {
    const __bf16* Xt = P.Xb + (size_t)t * Cdim;
    v8f acc[2][4] = {};
    gemm_seg32(acc, Xt,   (long)Tdim * Cdim, P.Wxh, Cdim, m0, n0, Cdim, lane);
    gemm_seg32(acc, P.RH, Cdim,              P.Whh, Cdim, m0, n0, Cdim, lane);

    EPILOGUE_2x4({
        const long idx = (long)row * Cdim + col;
        float ht = tanhf(accv + P.bxh[col] + P.bhh[col]);
        float z = P.Z[idx];
        float hn = z * P.Hf[idx] + (1.0f - z) * ht;
        P.Hf[idx] = hn;
        P.Hb[idx] = (__bf16)hn;
    })
}

// y job: Y[:,t,:] = H@Whq^T + bhq
__device__ void job_y(const GruParams& P, int t, int m0, int n0, int lane) {
    v8f acc[2][4] = {};
    gemm_seg32(acc, P.Hb, Cdim, P.Whq, Cdim, m0, n0, Cdim, lane);
    float* Yt = P.Y + (size_t)t * Cdim;
    EPILOGUE_2x4({
        Yt[(long)row * Tdim * Cdim + col] = accv + P.bhq[col];
    })
}

// ------------------------------ phases --------------------------------------
// gates: 64 jobs = 8 m-tiles(32 rows) x 4 n-groups(256 cols) x 2 gates
__device__ __forceinline__ void phase_gates(const GruParams& P, int t,
                                            int wave, int lane) {
    for (int job = blockIdx.x; job < 64; job += gridDim.x) {
        int g = job & 1, n = (job >> 1) & 3, m = job >> 3;
        job_gate(P, t, g, m * 32, n * 256 + wave * 64, lane);
    }
}
// update / y: 32 jobs = 8 m-tiles x 4 n-groups
__device__ __forceinline__ void phase_update(const GruParams& P, int t,
                                             int wave, int lane) {
    for (int job = blockIdx.x; job < 32; job += gridDim.x)
        job_update(P, t, (job >> 2) * 32, (job & 3) * 256 + wave * 64, lane);
}
__device__ __forceinline__ void phase_y(const GruParams& P, int t,
                                        int wave, int lane) {
    for (int job = blockIdx.x; job < 32; job += gridDim.x)
        job_y(P, t, (job >> 2) * 32, (job & 3) * 256 + wave * 64, lane);
}

// -------------------------- persistent kernel -------------------------------
__global__ void gru_persistent(GruParams P) {
    const int lane = threadIdx.x & 31;
    const int wave = threadIdx.x >> 5;

    phase_gates(P, 0, wave, lane);
    gbar(P);
    for (int t = 0; t < Tdim; ++t) {
        phase_update(P, t, wave, lane);         // needs gates(t)
        gbar(P);
        phase_y(P, t, wave, lane);              // needs update(t)
        if (t + 1 < Tdim)
            phase_gates(P, t + 1, wave, lane);  // also needs update(t); no hazard with y
        gbar(P);                                // protects Hb/Z/RH for next update
    }
}

// ------------------------------ prep kernels --------------------------------
__global__ void f2bf(const float* __restrict__ src, __bf16* __restrict__ dst,
                     long n) {
    long i = (long)blockIdx.x * blockDim.x + threadIdx.x;
    long stride = (long)gridDim.x * blockDim.x;
    for (; i < n; i += stride) dst[i] = (__bf16)src[i];
}

__global__ void init_h(const float* __restrict__ H0, float* __restrict__ Hf,
                       __bf16* __restrict__ Hb, long n,
                       unsigned* bar_cnt, unsigned* bar_gen) {
    if (blockIdx.x == 0 && threadIdx.x == 0) { *bar_cnt = 0u; *bar_gen = 0u; }
    long i = (long)blockIdx.x * blockDim.x + threadIdx.x;
    long stride = (long)gridDim.x * blockDim.x;
    for (; i < n; i += stride) {
        float v = H0[i];
        Hf[i] = v;
        Hb[i] = (__bf16)v;
    }
}

// ------------------------------- launcher -----------------------------------
extern "C" void kernel_launch(void* const* d_in, const int* in_sizes, int n_in,
                              void* d_out, int out_size, void* d_ws, size_t ws_size,
                              hipStream_t stream) {
    const float* X  = (const float*)d_in[0];
    const float* H0 = (const float*)d_in[1];
    const float* Wf[7] = { (const float*)d_in[2],  (const float*)d_in[4],
                           (const float*)d_in[6],  (const float*)d_in[8],
                           (const float*)d_in[10], (const float*)d_in[12],
                           (const float*)d_in[14] };       // Wxz Whz Wxr Whr Wxh Whh Whq

    // ---- workspace carve-out (all re-initialized every call) ----
    char* ws = (char*)d_ws;
    __bf16* Xb = (__bf16*)ws;            ws += (size_t)Bdim * Tdim * Cdim * 2;  // 128 MB
    __bf16* Wb[7];
    for (int i = 0; i < 7; ++i) { Wb[i] = (__bf16*)ws; ws += (size_t)Cdim * Cdim * 2; }
    float*  Hf = (float*)ws;             ws += (size_t)Bdim * Cdim * 4;
    __bf16* Hb = (__bf16*)ws;            ws += (size_t)Bdim * Cdim * 2;
    float*  Z  = (float*)ws;             ws += (size_t)Bdim * Cdim * 4;
    __bf16* RH = (__bf16*)ws;            ws += (size_t)Bdim * Cdim * 2;
    unsigned* bar = (unsigned*)ws;       ws += 256;       // cnt, gen

    // ---- per-call conversions to bf16 + state/barrier init ----
    f2bf<<<4096, 256, 0, stream>>>(X, Xb, (long)Bdim * Tdim * Cdim);
    for (int i = 0; i < 7; ++i)
        f2bf<<<512, 256, 0, stream>>>(Wf[i], Wb[i], (long)Cdim * Cdim);
    init_h<<<1024, 256, 0, stream>>>(H0, Hf, Hb, (long)Bdim * Cdim,
                                     bar, bar + 1);

    // ---- one persistent kernel for the whole recurrence ----
    GruParams P;
    P.Xb = Xb;
    P.Wxz = Wb[0]; P.Whz = Wb[1]; P.Wxr = Wb[2]; P.Whr = Wb[3];
    P.Wxh = Wb[4]; P.Whh = Wb[5]; P.Whq = Wb[6];
    P.bxz = (const float*)d_in[3];  P.bhz = (const float*)d_in[5];
    P.bxr = (const float*)d_in[7];  P.bhr = (const float*)d_in[9];
    P.bxh = (const float*)d_in[11]; P.bhh = (const float*)d_in[13];
    P.bhq = (const float*)d_in[15];
    P.Hf = Hf; P.Hb = Hb; P.Z = Z; P.RH = RH;
    P.Y = (float*)d_out;
    P.bar_cnt = bar; P.bar_gen = bar + 1;

    gru_persistent<<<NBLOCKS, NTHREADS, 0, stream>>>(P);

    (void)in_sizes; (void)n_in; (void)out_size; (void)ws_size;
}
